// AAD_Net_20169166422138
// MI455X (gfx1250) — matrix operand order repaired and checked by reading, not compiled, not run on hardware
//
#include <hip/hip_runtime.h>

// ---------------------------------------------------------------------------
// AAD-Net forward for MI455X (gfx1250, wave32). Compile-only environment.
// GEMM-shaped stages (Pearson-correlation einsum, FC1) use
// v_wmma_f32_16x16x32_f16 (f16 inputs, f32 accumulate). Everything else is
// VALU with LDS staging. Total work ~2.2 GFLOP / ~50MB traffic -> latency
// bound; design minimizes kernel count and recompute.
// ---------------------------------------------------------------------------

typedef __attribute__((ext_vector_type(16))) _Float16 v16h;
typedef __attribute__((ext_vector_type(8)))  float    v8f;

union V16H { v16h v; _Float16 f[16]; unsigned u[8]; };

#define TT 2048
#define BB 16

__device__ __forceinline__ float block_reduce(float v, float* sm) {
    const int tid = threadIdx.x;
    sm[tid] = v;
    __syncthreads();
    for (int s = 128; s > 0; s >>= 1) {
        if (tid < s) sm[tid] += sm[tid + s];
        __syncthreads();
    }
    const float r = sm[0];
    __syncthreads();
    return r;
}

// -------- per-channel batch-norm statistics over (B,T): mean + rsqrt(var+eps)
__global__ void bn_stats_kernel(const float* __restrict__ x,
                                float* __restrict__ stats, int C) {
    const int c = blockIdx.x;
    const int tid = threadIdx.x;
    float s = 0.f, ss = 0.f;
    for (int b = 0; b < BB; ++b) {
        const float* p = x + ((size_t)b * C + c) * TT;
        for (int t = tid; t < TT; t += 256) { float v = p[t]; s += v; ss += v * v; }
    }
    __shared__ float sm[256];
    s  = block_reduce(s, sm);
    ss = block_reduce(ss, sm);
    if (tid == 0) {
        const float n = (float)(BB * TT);
        const float m = s / n;
        const float var = ss / n - m * m;       // biased variance (matches ref)
        stats[2 * c]     = m;
        stats[2 * c + 1] = rsqrtf(var + 1e-5f);
    }
}

// -------- inception stage 1: BN-apply + all 1x1 convs (transform 32,
// compress 16/8/4/2, maxpool3+1x1 -> 8). Writes transform/pool channels into
// the 72-ch concat buffer Y and compressed channels into H(B,30,T).
__global__ void compress_kernel(const float* __restrict__ x,
                                const float* __restrict__ stats,
                                const float* __restrict__ gamma,
                                const float* __restrict__ beta,
                                const float* tw,  const float* tbias,
                                const float* cw0, const float* cb0,
                                const float* cw1, const float* cb1,
                                const float* cw2, const float* cb2,
                                const float* cw3, const float* cb3,
                                const float* pw,  const float* pb,
                                float* __restrict__ Y, float* __restrict__ H,
                                int in_c) {
    __shared__ float sw[70 * 64];
    __shared__ float snorm[64 * 4];
    const int tid = threadIdx.x;
    for (int i = tid; i < 70 * in_c; i += 256) {
        const int ci = i / 70, oc = i % 70;
        float w;
        if      (oc < 32) w = tw [(oc)      * in_c + ci];
        else if (oc < 48) w = cw0[(oc - 32) * in_c + ci];
        else if (oc < 56) w = cw1[(oc - 48) * in_c + ci];
        else if (oc < 60) w = cw2[(oc - 56) * in_c + ci];
        else if (oc < 62) w = cw3[(oc - 60) * in_c + ci];
        else              w = pw [(oc - 62) * in_c + ci];
        sw[ci * 70 + oc] = w;
    }
    for (int ci = tid; ci < in_c; ci += 256) {
        snorm[ci * 4 + 0] = stats[2 * ci];
        snorm[ci * 4 + 1] = stats[2 * ci + 1];
        snorm[ci * 4 + 2] = gamma[ci];
        snorm[ci * 4 + 3] = beta[ci];
    }
    __syncthreads();

    const int idx = blockIdx.x * 256 + tid;     // 128 blocks * 256 = B*T
    const int b = idx >> 11;
    const int t = idx & (TT - 1);

    float acc[70];
#pragma unroll
    for (int i = 0; i < 70; ++i) acc[i] = 0.f;

    for (int ci = 0; ci < in_c; ++ci) {
        const float* xp = x + ((size_t)b * in_c + ci) * TT;
        const float m  = snorm[ci * 4 + 0];
        const float r  = snorm[ci * 4 + 1];
        const float g  = snorm[ci * 4 + 2];
        const float be = snorm[ci * 4 + 3];
        const float xc = g * (xp[t] - m) * r + be;
        float pv = xc;                              // maxpool3, pad=1
        if (t > 0)      pv = fmaxf(pv, g * (xp[t - 1] - m) * r + be);
        if (t < TT - 1) pv = fmaxf(pv, g * (xp[t + 1] - m) * r + be);
        const float* wr = sw + ci * 70;
#pragma unroll
        for (int oc = 0; oc < 62; ++oc) acc[oc] += wr[oc] * xc;
#pragma unroll
        for (int oc = 62; oc < 70; ++oc) acc[oc] += wr[oc] * pv;
    }
    // transform -> concat channels 0..31
#pragma unroll
    for (int oc = 0; oc < 32; ++oc)
        Y[((size_t)b * 72 + oc) * TT + t] = acc[oc] + tbias[oc];
    // compressed -> H channels 0..29
#pragma unroll
    for (int j = 0; j < 16; ++j) H[((size_t)b * 30 + j)      * TT + t] = acc[32 + j] + cb0[j];
#pragma unroll
    for (int j = 0; j < 8;  ++j) H[((size_t)b * 30 + 16 + j) * TT + t] = acc[48 + j] + cb1[j];
#pragma unroll
    for (int j = 0; j < 4;  ++j) H[((size_t)b * 30 + 24 + j) * TT + t] = acc[56 + j] + cb2[j];
#pragma unroll
    for (int j = 0; j < 2;  ++j) H[((size_t)b * 30 + 28 + j) * TT + t] = acc[60 + j] + cb3[j];
    // pool branch -> concat channels 64..71
#pragma unroll
    for (int j = 0; j < 8;  ++j)
        Y[((size_t)b * 72 + 64 + j) * TT + t] = acc[62 + j] + pb[j];
}

// -------- inception stage 2: 'same' temporal convs from LDS-staged H tile
template <int C, int K, int WOFF, int HOFF, int OBASE>
__device__ __forceinline__ void branch_conv(const float* shH, const float* shW,
                                            const float* fb, float* Y,
                                            int b, int t) {
    float acc[8];
#pragma unroll
    for (int oc = 0; oc < 8; ++oc) acc[oc] = fb[oc];
    const int base = (int)threadIdx.x + 19 - (K - 1) / 2;   // halo = 19 (max pad)
    for (int c = 0; c < C; ++c) {
        const float* hrow = shH + (HOFF + c) * 166 + base;
#pragma unroll
        for (int k = 0; k < K; ++k) {
            const float hv = hrow[k];
#pragma unroll
            for (int oc = 0; oc < 8; ++oc)
                acc[oc] += shW[WOFF + (oc * C + c) * K + k] * hv;
        }
    }
#pragma unroll
    for (int oc = 0; oc < 8; ++oc)
        Y[((size_t)b * 72 + OBASE + oc) * TT + t] = acc[oc];
}

__global__ void featconv_kernel(const float* __restrict__ H,
                                const float* fw0, const float* fb0,
                                const float* fw1, const float* fb1,
                                const float* fw2, const float* fb2,
                                const float* fw3, const float* fb3,
                                float* __restrict__ Y) {
    __shared__ float shH[30 * 166];   // tile 128 + 2*19 halo
    __shared__ float shW[5712];
    const int b = blockIdx.x;
    const int tstart = blockIdx.y * 128;
    const int tid = threadIdx.x;
    for (int i = tid; i < 5712; i += 128) {
        float w;
        if      (i < 2432) w = fw0[i];
        else if (i < 4032) w = fw1[i - 2432];
        else if (i < 5088) w = fw2[i - 4032];
        else               w = fw3[i - 5088];
        shW[i] = w;
    }
    for (int i = tid; i < 30 * 166; i += 128) {
        const int c = i / 166, p = i % 166;
        const int gt = tstart - 19 + p;
        shH[i] = (gt >= 0 && gt < TT) ? H[((size_t)b * 30 + c) * TT + gt] : 0.f;
    }
    __syncthreads();
    const int t = tstart + tid;
    branch_conv<16, 19, 0,    0,  32>(shH, shW, fb0, Y, b, t);
    branch_conv<8,  25, 2432, 16, 40>(shH, shW, fb1, Y, b, t);
    branch_conv<4,  33, 4032, 24, 48>(shH, shW, fb2, Y, b, t);
    branch_conv<2,  39, 5088, 28, 56>(shH, shW, fb3, Y, b, t);
}

// -------- fold out-BN affine, subtract per-(b,ch) mean, emit f16 devs + f32 ss
__global__ void dev_kernel(const float* __restrict__ Y,
                           const float* __restrict__ obn,
                           const float* __restrict__ og,
                           const float* __restrict__ ob,
                           _Float16* __restrict__ dev,
                           float* __restrict__ ssout,
                           int dstC, int choff) {
    const int b = blockIdx.x / 72, c = blockIdx.x % 72;
    const int tid = threadIdx.x;
    const float* src = Y + ((size_t)b * 72 + c) * TT;
    const float scale = og[c] * obn[2 * c + 1];
    const float shift = ob[c] - og[c] * obn[2 * c] * obn[2 * c + 1];
    float s = 0.f;
    for (int t = tid; t < TT; t += 256) s += scale * src[t] + shift;
    __shared__ float sm[256];
    s = block_reduce(s, sm);
    const float mean = s * (1.f / (float)TT);
    _Float16* drow = dev + ((size_t)b * dstC + choff + c) * TT;
    float ss = 0.f;
    for (int t = tid; t < TT; t += 256) {
        const float d = scale * src[t] + shift - mean;
        ss += d * d;                       // f32 sumsq (denominator stays exact-ish)
        drow[t] = (_Float16)d;
    }
    ss = block_reduce(ss, sm);
    if (tid == 0) ssout[b * dstC + choff + c] = ss;
}

__global__ void pad_zero_kernel(_Float16* __restrict__ E, float* __restrict__ ssE) {
    const int i = blockIdx.x * 256 + threadIdx.x;   // B*8*T elements
    if (i < BB * 8 * TT) {
        const int b = i / (8 * TT);
        const int rem = i % (8 * TT);
        const int c = 72 + rem / TT;
        const int t = rem % TT;
        E[((size_t)b * 80 + c) * TT + t] = (_Float16)0.f;
    }
    if (i < BB * 8) ssE[(i / 8) * 80 + 72 + (i % 8)] = 0.f;
}

// -------- Pearson correlation: per-batch (144x2048)x(2048x72) via WMMA f16
// One wave per 16x16 tile; K loop 2048/32. A/B/D VGPR layouts per ISA 7.12.2.
__global__ void corr_wmma_kernel(const _Float16* __restrict__ S,   // (B,144,T)
                                 const _Float16* __restrict__ E,   // (B,80,T) padded
                                 const float* __restrict__ ssS,
                                 const float* __restrict__ ssE,
                                 float* __restrict__ coefs) {
    const int lane = threadIdx.x;
    const int b  = blockIdx.x;
    const int s0 = blockIdx.y * 16;
    const int e0 = blockIdx.z * 16;
    const int half = lane >> 4, ml = lane & 15;

    const _Float16* Arow = S + ((size_t)(b * 144 + s0 + ml)) * TT;  // A row = M
    const _Float16* Brow = E + ((size_t)(b * 80 + e0 + ml)) * TT;   // B col = N

    v8f c = {};
    for (int kk = 0; kk < TT / 32; ++kk) {
        const int tb = kk * 32;
        V16H A, Bm;
#pragma unroll
        for (int v = 0; v < 8; ++v) {       // A 16x32: K pair per VGPR
            const int k0 = (v >> 2) * 16 + half * 8 + (v & 3) * 2;
            A.u[v] = *(const unsigned*)(Arow + tb + k0);
        }
#pragma unroll
        for (int v = 0; v < 8; ++v) {       // B 32x16: K pair per VGPR
            const int k0 = (v >> 2) * 16 + half * 8 + (v & 3) * 2;
            Bm.u[v] = *(const unsigned*)(Brow + tb + k0);
        }
        c = __builtin_amdgcn_wmma_f32_16x16x32_f16(false, A.v, false, Bm.v,
                                                   (short)0, c, false, false);
    }
    const int e = e0 + ml;                  // D: N = lane&15, M = r + 8*(lane>>4)
    if (e < 72) {
        const float se = ssE[b * 80 + e];
#pragma unroll
        for (int r = 0; r < 8; ++r) {
            const int s = s0 + r + 8 * half;
            const float den = sqrtf(ssS[b * 144 + s] * se) + 1e-8f;
            coefs[(size_t)b * 10368 + s * 72 + e] = c[r] / den;
        }
    }
}

// -------- FC1: (16 x 10368) @ (10368 x 256), M=16 == batch, WMMA + bias + ELU
__global__ void fc1_wmma_kernel(const float* __restrict__ coefs,
                                const float* __restrict__ w1,
                                const float* __restrict__ b1,
                                float* __restrict__ h1) {
    const int lane = threadIdx.x;
    const int n0 = blockIdx.x * 16;
    const int half = lane >> 4, ml = lane & 15;
    const float* Arow = coefs + (size_t)ml * 10368;
    const int kOff = ml + half * 16;
    v8f c = {};
    for (int kk = 0; kk < 10368 / 32; ++kk) {
        const int tb = kk * 32;
        V16H A, Bm;
#pragma unroll
        for (int v = 0; v < 8; ++v) {
            const int k0 = (v >> 2) * 16 + half * 8 + (v & 3) * 2;
            A.f[2 * v]     = (_Float16)Arow[tb + k0];
            A.f[2 * v + 1] = (_Float16)Arow[tb + k0 + 1];
        }
        const float* wcol = w1 + (size_t)tb * 256 + n0 + ml;
#pragma unroll
        for (int v = 0; v < 8; ++v) {
            const int k0 = (v >> 2) * 16 + half * 8 + (v & 3) * 2;
            Bm.f[2 * v]     = (_Float16)wcol[(size_t)k0 * 256];
            Bm.f[2 * v + 1] = (_Float16)wcol[(size_t)(k0 + 1) * 256];
        }
        c = __builtin_amdgcn_wmma_f32_16x16x32_f16(false, A.v, false, Bm.v,
                                                   (short)0, c, false, false);
    }
    const int n = n0 + ml;
    const float bias = b1[n];
#pragma unroll
    for (int r = 0; r < 8; ++r) {
        const int m = r + 8 * half;        // batch index
        const float v = c[r] + bias;
        h1[m * 256 + n] = v > 0.f ? v : expm1f(v);   // ELU(alpha=1)
    }
}

// -------- FC2 + softmax (16 x 256 -> 16 x 2)
__global__ void fc2_softmax_kernel(const float* __restrict__ h1,
                                   const float* __restrict__ w2,
                                   const float* __restrict__ b2,
                                   float* __restrict__ out) {
    const int b = threadIdx.x;
    if (b < BB) {
        float l0 = b2[0], l1 = b2[1];
        for (int n = 0; n < 256; ++n) {
            const float h = h1[b * 256 + n];
            l0 += h * w2[n * 2 + 0];
            l1 += h * w2[n * 2 + 1];
        }
        const float mx = fmaxf(l0, l1);
        const float e0 = expf(l0 - mx), e1 = expf(l1 - mx);
        const float inv = 1.f / (e0 + e1);
        out[b * 2 + 0] = e0 * inv;
        out[b * 2 + 1] = e1 * inv;
    }
}

// ---------------------------------------------------------------------------
// Input leaf order assumption: top-level in setup_inputs() insertion order
// (eeg, stima, stimb, eeg_params, stim_params, clsf_params); nested dicts
// flattened with JAX's sorted-key pytree convention, lists in order.
// Per inception-params block (24 leaves):
//   +0..3  comp_b[0..3]   +4..7  comp_w[0..3]   +8..11 feat_b[0..3]
//   +12..15 feat_w[0..3]  +16 in_beta  +17 in_gamma  +18 out_beta
//   +19 out_gamma  +20 pool_b  +21 pool_w  +22 transform_b  +23 transform_w
// clsf (sorted): +0 b1, +1 b2, +2 w1, +3 w2.
// ---------------------------------------------------------------------------
extern "C" void kernel_launch(void* const* d_in, const int* in_sizes, int n_in,
                              void* d_out, int out_size, void* d_ws, size_t ws_size,
                              hipStream_t stream) {
    (void)in_sizes; (void)n_in; (void)out_size; (void)ws_size;
    auto P = [&](int i) { return (const float*)d_in[i]; };
    const float* eeg   = P(0);
    const float* stima = P(1);
    const float* stimb = P(2);
    const int EP = 3, SP = 27, CP = 51;

    float* ws        = (float*)d_ws;
    float* stats_eeg = ws;               // 128
    float* stats_sa  = ws + 128;         // 2
    float* stats_sb  = ws + 130;         // 2
    float* obn_eeg   = ws + 132;         // 144
    float* obn_sa    = ws + 276;         // 144
    float* obn_sb    = ws + 420;         // 144
    float* ssE       = ws + 564;         // B*80  = 1280
    float* ssS       = ws + 1844;        // B*144 = 2304
    float* h1        = ws + 4148;        // B*256 = 4096
    float* coefs     = ws + 8244;        // B*10368 = 165888
    float* Y_eeg     = ws + 174132;      // B*72*T = 2359296
    float* Y_sa      = ws + 2533428;
    float* Y_sb      = ws + 4892724;
    float* Hbuf      = ws + 7252020;     // B*30*T = 983040 (reused)
    _Float16* Edev   = (_Float16*)(ws + 8235060);   // B*80*T halfs
    _Float16* Sdev   = (_Float16*)(ws + 9545780);   // B*144*T halfs
    // total ~11.9M floats (~48 MB) of d_ws

    auto run_inception = [&](const float* x, int in_c, int pb,
                             float* stats, float* Y, float* obn) {
        bn_stats_kernel<<<in_c, 256, 0, stream>>>(x, stats, in_c);
        compress_kernel<<<128, 256, 0, stream>>>(
            x, stats, P(pb + 17), P(pb + 16),
            P(pb + 23), P(pb + 22),
            P(pb + 4), P(pb + 0), P(pb + 5), P(pb + 1),
            P(pb + 6), P(pb + 2), P(pb + 7), P(pb + 3),
            P(pb + 21), P(pb + 20), Y, Hbuf, in_c);
        featconv_kernel<<<dim3(BB, TT / 128), 128, 0, stream>>>(
            Hbuf,
            P(pb + 12), P(pb + 8),  P(pb + 13), P(pb + 9),
            P(pb + 14), P(pb + 10), P(pb + 15), P(pb + 11), Y);
        bn_stats_kernel<<<72, 256, 0, stream>>>(Y, obn, 72);
    };

    // EEG encoder -> Edev (padded 72->80 channels)
    run_inception(eeg, 64, EP, stats_eeg, Y_eeg, obn_eeg);
    dev_kernel<<<BB * 72, 256, 0, stream>>>(Y_eeg, obn_eeg, P(EP + 19), P(EP + 18),
                                            Edev, ssE, 80, 0);
    pad_zero_kernel<<<(BB * 8 * TT) / 256, 256, 0, stream>>>(Edev, ssE);

    // Stim encoders (shared params, independent batch stats) -> Sdev (144 ch)
    run_inception(stima, 1, SP, stats_sa, Y_sa, obn_sa);
    dev_kernel<<<BB * 72, 256, 0, stream>>>(Y_sa, obn_sa, P(SP + 19), P(SP + 18),
                                            Sdev, ssS, 144, 0);
    run_inception(stimb, 1, SP, stats_sb, Y_sb, obn_sb);
    dev_kernel<<<BB * 72, 256, 0, stream>>>(Y_sb, obn_sb, P(SP + 19), P(SP + 18),
                                            Sdev, ssS, 144, 72);

    // Pearson correlation (WMMA) -> coefs (16,10368)
    corr_wmma_kernel<<<dim3(BB, 144 / 16, 80 / 16), 32, 0, stream>>>(
        Sdev, Edev, ssS, ssE, coefs);

    // Classifier
    fc1_wmma_kernel<<<256 / 16, 32, 0, stream>>>(coefs, P(CP + 2), P(CP + 0), h1);
    fc2_softmax_kernel<<<1, 32, 0, stream>>>(h1, P(CP + 3), P(CP + 1), (float*)d_out);
}
